// StackedAttentionGRU_49873160241524
// MI455X (gfx1250) — compile-verified, matrix operand order checked by reading, not run on hardware
//
#include <hip/hip_runtime.h>

// ---------------------------------------------------------------------------
// StackedAttentionGRU for MI455X (gfx1250): bf16 WMMA GEMMs + f32 elementwise
// B=256, T=128, S=128, H=1024, NL=2
// ---------------------------------------------------------------------------

typedef __bf16 bf16_t;
typedef __attribute__((ext_vector_type(16))) __bf16 v16bf;
typedef __attribute__((ext_vector_type(8)))  __bf16 v8bf;
typedef __attribute__((ext_vector_type(8)))  float  v8f;

#define GB 256
#define GT 128
#define GS 128
#define GH 1024

// ---------------------------------------------------------------------------
// Generic WMMA GEMM:  out = act( A(MxK,bf16) @ W(NxK,bf16)^T + bias )
// A is split: rows come from A0 (k < ksplit) and A1 (k >= ksplit) to support
// the [x_t | h] and [wctx | hy] concatenations without materializing them.
// Each wave computes a 32x32 C macro-tile as 2x2 WMMA tiles: per K-step it
// loads 2 A-fragments + 2 B-fragments (8x global_load_b128) and issues 4
// v_wmma_f32_16x16x32_bf16 -> 2 loads per WMMA (register-level reuse).
// Epilogue can fan out to up to 2 f32 and 2 bf16 destinations.
// Requires: M % 32 == 0, N % 256 == 0, K % 32 == 0, ksplit % 32 == 0.
// ---------------------------------------------------------------------------
__global__ __launch_bounds__(256)
void wmma_gemm_kernel(const bf16_t* __restrict__ A0, int lda0,
                      const bf16_t* __restrict__ A1, int lda1, int ksplit,
                      const bf16_t* __restrict__ W,  int ldw,
                      const float*  __restrict__ bias,
                      int M, int N, int K,
                      float*  outF,  int ldoF,
                      bf16_t* outB,  int ldoB,
                      bf16_t* outB2, int ldoB2,
                      float*  outF2, int ldoF2,
                      int act)  // 0 = none, 1 = tanh
{
    const int lane = threadIdx.x & 31;
    const int wave = threadIdx.x >> 5;
    const int m0 = blockIdx.x * 32;
    const int n0 = (blockIdx.y * 8 + wave) * 32;
    if (n0 >= N || m0 >= M) return;  // wave-uniform: EXEC stays all-ones

    const int ln = lane & 15;
    const int hi = (lane >> 4) << 3;  // 0 or 8 (ISA 16-bit A/B fragment layout)

    // Fragment row pointers (two 16-row tiles each for A and W)
    const bf16_t* a0lo = A0 + (size_t)(m0 + ln) * lda0;
    const bf16_t* a0hi = A0 + (size_t)(m0 + 16 + ln) * lda0;
    const bf16_t* a1lo = A1 + (size_t)(m0 + ln) * lda1;
    const bf16_t* a1hi = A1 + (size_t)(m0 + 16 + ln) * lda1;
    const bf16_t* wp0  = W  + (size_t)(n0 + ln) * ldw;
    const bf16_t* wp1  = W  + (size_t)(n0 + 16 + ln) * ldw;

    auto ldfrag = [&](const bf16_t* base) -> v16bf {
        v16bf f;
        ((v8bf*)&f)[0] = *(const v8bf*)(base + hi);
        ((v8bf*)&f)[1] = *(const v8bf*)(base + 16 + hi);
        return f;
    };

    v8f acc00 = {}, acc01 = {}, acc10 = {}, acc11 = {};

    // k < ksplit : operand rows from A0
    #pragma unroll 2
    for (int k0 = 0; k0 < ksplit; k0 += 32) {
        v16bf a0 = ldfrag(a0lo + k0);
        v16bf a1 = ldfrag(a0hi + k0);
        v16bf b0 = ldfrag(wp0 + k0);
        v16bf b1 = ldfrag(wp1 + k0);
        acc00 = __builtin_amdgcn_wmma_f32_16x16x32_bf16(false, a0, false, b0, (short)0, acc00, false, false);
        acc01 = __builtin_amdgcn_wmma_f32_16x16x32_bf16(false, a0, false, b1, (short)0, acc01, false, false);
        acc10 = __builtin_amdgcn_wmma_f32_16x16x32_bf16(false, a1, false, b0, (short)0, acc10, false, false);
        acc11 = __builtin_amdgcn_wmma_f32_16x16x32_bf16(false, a1, false, b1, (short)0, acc11, false, false);
    }
    // k >= ksplit : operand rows from A1
    #pragma unroll 2
    for (int k0 = ksplit; k0 < K; k0 += 32) {
        const int ka = k0 - ksplit;
        v16bf a0 = ldfrag(a1lo + ka);
        v16bf a1 = ldfrag(a1hi + ka);
        v16bf b0 = ldfrag(wp0 + k0);
        v16bf b1 = ldfrag(wp1 + k0);
        acc00 = __builtin_amdgcn_wmma_f32_16x16x32_bf16(false, a0, false, b0, (short)0, acc00, false, false);
        acc01 = __builtin_amdgcn_wmma_f32_16x16x32_bf16(false, a0, false, b1, (short)0, acc01, false, false);
        acc10 = __builtin_amdgcn_wmma_f32_16x16x32_bf16(false, a1, false, b0, (short)0, acc10, false, false);
        acc11 = __builtin_amdgcn_wmma_f32_16x16x32_bf16(false, a1, false, b1, (short)0, acc11, false, false);
    }

    // Epilogue. C layout: VGPR j -> row tile_m + j + ((lane>>4)<<3), col tile_n + (lane&15)
    const float bv0 = bias ? bias[n0 + ln] : 0.0f;
    const float bv1 = bias ? bias[n0 + 16 + ln] : 0.0f;
    const int mb = (lane >> 4) << 3;
    v8f* accs[4] = { &acc00, &acc01, &acc10, &acc11 };
    #pragma unroll
    for (int r = 0; r < 2; ++r) {
        #pragma unroll
        for (int c = 0; c < 2; ++c) {
            v8f& a = *accs[r * 2 + c];
            const int n = n0 + 16 * c + ln;
            const float bvv = c ? bv1 : bv0;
            #pragma unroll
            for (int j = 0; j < 8; ++j) {
                float v = a[j] + bvv;
                if (act == 1) v = tanhf(v);
                const int m = m0 + 16 * r + mb + j;
                if (outF)  outF [(size_t)m * ldoF  + n] = v;
                if (outF2) outF2[(size_t)m * ldoF2 + n] = v;
                const bf16_t vb = (bf16_t)v;
                if (outB)  outB [(size_t)m * ldoB  + n] = vb;
                if (outB2) outB2[(size_t)m * ldoB2 + n] = vb;
            }
        }
    }
}

// ---------------------------------------------------------------------------
// GRU gate fusion: pre = (B x 4H) [r_pre | z_pre | i_n | h_n]
// hy = (1-z)*tanh(i_n + r*h_n) + z*h_prev ; writes hy f32 (in-place over
// hstate is safe: each thread reads only its own element before writing) and
// bf16 hy into cat2[:,H:].
// ---------------------------------------------------------------------------
__global__ __launch_bounds__(256)
void gate_fuse_kernel(const float* __restrict__ pre,
                      float* __restrict__ hstate,
                      bf16_t* __restrict__ cat2)
{
    const int idx = blockIdx.x * blockDim.x + threadIdx.x;  // over B*H
    const int b = idx >> 10;          // H = 1024
    const int h = idx & (GH - 1);
    const float* p = pre + (size_t)b * 4 * GH;
    const float r = 1.0f / (1.0f + __expf(-p[h]));
    const float z = 1.0f / (1.0f + __expf(-p[GH + h]));
    const float n = tanhf(p[2 * GH + h] + r * p[3 * GH + h]);
    const float hp = hstate[(size_t)b * GH + h];
    const float v = (1.0f - z) * n + z * hp;
    hstate[(size_t)b * GH + h] = v;
    cat2[(size_t)b * 2 * GH + GH + h] = (bf16_t)v;
}

// ---------------------------------------------------------------------------
// Soft-dot attention, one block per batch element b:
//   scores[s] = <ctx[s,b,:], target[b,:]>, softmax over s, wctx = attn @ ctx
// writes wctx (bf16) into cat2[:, 0:H]
// ---------------------------------------------------------------------------
__global__ __launch_bounds__(256)
void attention_kernel(const float* __restrict__ ctx,
                      const float* __restrict__ target,
                      bf16_t* __restrict__ cat2)
{
    __shared__ float tgt[GH];
    __shared__ float sc[GS];
    const int b = blockIdx.x;
    const int tid = threadIdx.x;
    const int lane = tid & 31, wave = tid >> 5;

    for (int i = tid; i < GH; i += 256) tgt[i] = target[(size_t)b * GH + i];
    __syncthreads();

    for (int s = wave; s < GS; s += 8) {
        const float* row = ctx + (size_t)s * GB * GH + (size_t)b * GH;
        float p = 0.0f;
        for (int h = lane; h < GH; h += 32) p += row[h] * tgt[h];
        for (int off = 16; off; off >>= 1) p += __shfl_xor(p, off);
        if (lane == 0) sc[s] = p;
    }
    __syncthreads();

    if (wave == 0) {
        float m = -3.0e38f;
        for (int s = lane; s < GS; s += 32) m = fmaxf(m, sc[s]);
        for (int off = 16; off; off >>= 1) m = fmaxf(m, __shfl_xor(m, off));
        float sum = 0.0f;
        for (int s = lane; s < GS; s += 32) {
            float e = __expf(sc[s] - m);
            sc[s] = e;
            sum += e;
        }
        for (int off = 16; off; off >>= 1) sum += __shfl_xor(sum, off);
        const float inv = 1.0f / sum;
        for (int s = lane; s < GS; s += 32) sc[s] *= inv;
    }
    __syncthreads();

    float acc[4] = {0.f, 0.f, 0.f, 0.f};
    for (int s = 0; s < GS; ++s) {
        const float a = sc[s];
        const float* row = ctx + (size_t)s * GB * GH + (size_t)b * GH;
        #pragma unroll
        for (int j = 0; j < 4; ++j) acc[j] += a * row[tid + 256 * j];
    }
    #pragma unroll
    for (int j = 0; j < 4; ++j)
        cat2[(size_t)b * 2 * GH + tid + 256 * j] = (bf16_t)acc[j];
}

// ---------------------------------------------------------------------------
// Conversion / packing kernels (run every call; cheap vs. the scan)
// ---------------------------------------------------------------------------
__global__ __launch_bounds__(256)
void cvt_bf16_kernel(const float* __restrict__ src, bf16_t* __restrict__ dst, long n)
{
    long i = (long)blockIdx.x * blockDim.x + threadIdx.x;
    if (i < n) dst[i] = (bf16_t)src[i];
}

// Wg: (4H x 2H) bf16 = rows [ [Wih_rz|Whh_rz] ; [Wih_n|0] ; [0|Whh_n] ]
__global__ __launch_bounds__(256)
void pack_gates_w_kernel(const float* __restrict__ Wih, const float* __restrict__ Whh,
                         bf16_t* __restrict__ Wg)
{
    const long idx = (long)blockIdx.x * blockDim.x + threadIdx.x;  // 4H*2H
    const int r = (int)(idx / (2 * GH));
    const int c = (int)(idx % (2 * GH));
    float v = 0.0f;
    if (r < 2 * GH) {
        v = (c < GH) ? Wih[(size_t)r * GH + c] : Whh[(size_t)r * GH + (c - GH)];
    } else if (r < 3 * GH) {
        if (c < GH) v = Wih[(size_t)r * GH + c];                 // i_n rows (2H..3H)
    } else {
        if (c >= GH) v = Whh[(size_t)(r - GH) * GH + (c - GH)];  // h_n rows
    }
    Wg[idx] = (bf16_t)v;
}

// bg: (4H) f32 = [b_ih_r+b_hh_r, b_ih_z+b_hh_z, b_ih_n, b_hh_n]
__global__ __launch_bounds__(256)
void pack_gates_bias_kernel(const float* __restrict__ bih, const float* __restrict__ bhh,
                            float* __restrict__ bg)
{
    const int i = blockIdx.x * blockDim.x + threadIdx.x;  // 4H
    if (i < 2 * GH)       bg[i] = bih[i] + bhh[i];
    else if (i < 3 * GH)  bg[i] = bih[i];
    else                  bg[i] = bhh[i - GH];
}

// ---------------------------------------------------------------------------
// Host orchestration
// ---------------------------------------------------------------------------
static inline void* ws_alloc(char** p, size_t bytes)
{
    void* r = *p;
    *p += (bytes + 255) & ~(size_t)255;
    return r;
}

extern "C" void kernel_launch(void* const* d_in, const int* in_sizes, int n_in,
                              void* d_out, int out_size, void* d_ws, size_t ws_size,
                              hipStream_t stream)
{
    (void)in_sizes; (void)n_in; (void)out_size; (void)ws_size;

    const float* x   = (const float*)d_in[0];   // (B,T,H)
    const float* h0  = (const float*)d_in[1];   // (B,H)
    const float* ctx = (const float*)d_in[2];   // (S,B,H)

    const float* Wih[2],* Whh[2],* bih[2],* bhh[2],* Wain[2],* Waout[2];
    for (int l = 0; l < 2; ++l) {
        const int base = 3 + 6 * l;
        Wih[l]   = (const float*)d_in[base + 0];   // (3H,H)
        Whh[l]   = (const float*)d_in[base + 1];   // (3H,H)
        bih[l]   = (const float*)d_in[base + 2];   // (3H)
        bhh[l]   = (const float*)d_in[base + 3];   // (3H)
        Wain[l]  = (const float*)d_in[base + 4];   // (H,H)
        Waout[l] = (const float*)d_in[base + 5];   // (H,2H)
    }
    float* out = (float*)d_out;   // (B,T,H) then (2,B,H)

    // ---- workspace carve-up ----
    char* p = (char*)d_ws;
    bf16_t* xbuf[2];
    xbuf[0] = (bf16_t*)ws_alloc(&p, (size_t)GB * GT * GH * sizeof(bf16_t));
    xbuf[1] = (bf16_t*)ws_alloc(&p, (size_t)GB * GT * GH * sizeof(bf16_t));
    bf16_t* WgB[2]; float* bgB[2]; bf16_t* WainB[2]; bf16_t* WaoutB[2];
    for (int l = 0; l < 2; ++l) {
        WgB[l]    = (bf16_t*)ws_alloc(&p, (size_t)4 * GH * 2 * GH * sizeof(bf16_t));
        bgB[l]    = (float*) ws_alloc(&p, (size_t)4 * GH * sizeof(float));
        WainB[l]  = (bf16_t*)ws_alloc(&p, (size_t)GH * GH * sizeof(bf16_t));
        WaoutB[l] = (bf16_t*)ws_alloc(&p, (size_t)GH * 2 * GH * sizeof(bf16_t));
    }
    float*  pre    = (float*) ws_alloc(&p, (size_t)GB * 4 * GH * sizeof(float));
    float*  hstate = (float*) ws_alloc(&p, (size_t)GB * GH * sizeof(float));
    bf16_t* hbf    = (bf16_t*)ws_alloc(&p, (size_t)GB * GH * sizeof(bf16_t));
    float*  tgtbuf = (float*) ws_alloc(&p, (size_t)GB * GH * sizeof(float));
    bf16_t* cat2   = (bf16_t*)ws_alloc(&p, (size_t)GB * 2 * GH * sizeof(bf16_t));

    // ---- one-time conversions / packs (per call) ----
    {
        long n = (long)GB * GT * GH;
        cvt_bf16_kernel<<<(unsigned)((n + 255) / 256), 256, 0, stream>>>(x, xbuf[0], n);
    }
    for (int l = 0; l < 2; ++l) {
        long nw = (long)4 * GH * 2 * GH;
        pack_gates_w_kernel<<<(unsigned)((nw + 255) / 256), 256, 0, stream>>>(Wih[l], Whh[l], WgB[l]);
        pack_gates_bias_kernel<<<(4 * GH) / 256, 256, 0, stream>>>(bih[l], bhh[l], bgB[l]);
        long na = (long)GH * GH;
        cvt_bf16_kernel<<<(unsigned)((na + 255) / 256), 256, 0, stream>>>(Wain[l], WainB[l], na);
        long no = (long)GH * 2 * GH;
        cvt_bf16_kernel<<<(unsigned)((no + 255) / 256), 256, 0, stream>>>(Waout[l], WaoutB[l], no);
    }

    // ---- layers ----
    for (int l = 0; l < 2; ++l) {
        hipMemcpyAsync(hstate, h0, (size_t)GB * GH * sizeof(float),
                       hipMemcpyDeviceToDevice, stream);
        {
            long n = (long)GB * GH;
            cvt_bf16_kernel<<<(unsigned)((n + 255) / 256), 256, 0, stream>>>(h0, hbf, n);
        }

        for (int t = 0; t < GT; ++t) {
            // (1) fused gates GEMM: pre = [x_t|h] @ Wg^T + bg    (M=B,N=4H,K=2H)
            {
                dim3 grid(GB / 32, (4 * GH) / 256);
                wmma_gemm_kernel<<<grid, 256, 0, stream>>>(
                    xbuf[l] + (size_t)t * GH, GT * GH,   // A0: row b at (b*T+t)*H
                    hbf, GH, GH,                         // A1, ksplit=H
                    WgB[l], 2 * GH, bgB[l],
                    GB, 4 * GH, 2 * GH,
                    pre, 4 * GH,
                    nullptr, 0, nullptr, 0, nullptr, 0, 0);
            }
            // (2) gate fusion -> hy f32 (in-place over hstate), hy bf16 into cat2[:,H:]
            gate_fuse_kernel<<<(GB * GH) / 256, 256, 0, stream>>>(pre, hstate, cat2);
            // (3) target GEMM: target = hy_bf16 @ Wa_in^T  (M=B,N=H,K=H)
            {
                dim3 grid(GB / 32, GH / 256);
                wmma_gemm_kernel<<<grid, 256, 0, stream>>>(
                    cat2 + GH, 2 * GH,                   // A0 = hy slice of cat2
                    cat2 + GH, 2 * GH, GH,               // ksplit = K
                    WainB[l], GH, nullptr,
                    GB, GH, GH,
                    tgtbuf, GH,
                    nullptr, 0, nullptr, 0, nullptr, 0, 0);
            }
            // (4) attention -> wctx bf16 into cat2[:,0:H]
            attention_kernel<<<GB, 256, 0, stream>>>(ctx, tgtbuf, cat2);
            // (5) output GEMM: h' = tanh([wctx|hy] @ Wa_out^T); fan out
            {
                dim3 grid(GB / 32, GH / 256);
                wmma_gemm_kernel<<<grid, 256, 0, stream>>>(
                    cat2, 2 * GH,
                    cat2, 2 * GH, 2 * GH,                // ksplit = K (all from A0)
                    WaoutB[l], 2 * GH, nullptr,
                    GB, GH, 2 * GH,
                    hstate, GH,                          // f32 recurrent state
                    hbf, GH,                             // bf16 recurrent state
                    (l == 0) ? (xbuf[1] + (size_t)t * GH) : nullptr, GT * GH,
                    (l == 1) ? (out + (size_t)t * GH) : nullptr, GT * GH,
                    1 /*tanh*/);
            }
        }
        // h_fin[l] = final h
        hipMemcpyAsync(out + (size_t)GB * GT * GH + (size_t)l * GB * GH,
                       hstate, (size_t)GB * GH * sizeof(float),
                       hipMemcpyDeviceToDevice, stream);
    }
}
